// CausalMultiHeadAttentionWithCache_41059887350470
// MI455X (gfx1250) — matrix operand order repaired and checked by reading, not compiled, not run on hardware
//
#include <hip/hip_runtime.h>
#include <hip/hip_bf16.h>

// ---------------- problem constants ----------------
#define BB   2
#define TT   2048
#define CC   1024
#define HH   16
#define HD   64
#define MM   (BB * TT)      // 4096

// ---------------- types ----------------
typedef __attribute__((ext_vector_type(16))) __bf16 v16bf;
typedef __attribute__((ext_vector_type(8)))  __bf16 bf16x8;
typedef __attribute__((ext_vector_type(8)))  float  v8f;

static __device__ __forceinline__ __bf16 f2bf(float f) {
    unsigned int u = __builtin_bit_cast(unsigned int, f);
    unsigned int r = (u + 0x7FFFu + ((u >> 16) & 1u)) >> 16;
    return __builtin_bit_cast(__bf16, (unsigned short)r);
}

// Assemble a 16-element bf16 fragment from two 16-byte chunks.
static __device__ __forceinline__ v16bf frag_ld(const __bf16* p0, const __bf16* p1) {
    bf16x8 lo = *(const bf16x8*)p0;
    bf16x8 hi = *(const bf16x8*)p1;
    v16bf r;
#pragma unroll
    for (int i = 0; i < 8; ++i) { r[i] = lo[i]; r[i + 8] = hi[i]; }
    return r;
}

static __device__ __forceinline__ v8f wmma_bf16(v16bf a, v16bf b, v8f c) {
    return __builtin_amdgcn_wmma_f32_16x16x32_bf16(
        /*neg_a=*/false, a, /*neg_b=*/false, b,
        /*c_mod=*/(short)0, c, /*reuse_a=*/false, /*reuse_b=*/false);
}

// ---------------- fp32 -> bf16 convert (4 elems / thread) ----------------
__global__ __launch_bounds__(256) void cvt_f32_bf16(const float* __restrict__ in,
                                                    __bf16* __restrict__ out, int n4) {
    int i = blockIdx.x * 256 + threadIdx.x;
    if (i < n4) {
        const float4 f = ((const float4*)in)[i];
        __bf16* o = out + (size_t)i * 4;
        o[0] = f2bf(f.x); o[1] = f2bf(f.y); o[2] = f2bf(f.z); o[3] = f2bf(f.w);
    }
}

// ---------------- GEMM: D[m,n] = sum_k A[m,k] * W[n,k]  (y = x @ W^T) ------
// Block: 256 threads = 8 waves laid out 2(M) x 4(N); block tile 32 x 256.
// Each wave: 16x64 output slab, no LDS (192MB L2 holds everything).
// mode 0: store bf16 [B,H,T,hd] (Q layout)   mode 1: same (K layout)
// mode 2: store bf16 [B,H,hd,T] (V^T layout) mode 3: store f32 [M,C] + bias
__global__ __launch_bounds__(256) void gemm_bf16(const __bf16* __restrict__ A,
                                                 const __bf16* __restrict__ W,
                                                 __bf16* __restrict__ outb,
                                                 float*  __restrict__ outf,
                                                 const float* __restrict__ bias,
                                                 int mode) {
    const int lane  = threadIdx.x & 31;
    const int w     = threadIdx.x >> 5;
    const int mbase = blockIdx.y * 32 + (w >> 2) * 16;
    const int nbase = blockIdx.x * 256 + (w & 3) * 64;
    const int col   = lane & 15;
    const int half  = lane >> 4;

    v8f zero = {0.f,0.f,0.f,0.f,0.f,0.f,0.f,0.f};
    v8f acc[4] = {zero, zero, zero, zero};

    const __bf16* arow = A + (size_t)(mbase + col) * CC;   // A fragment row
    const __bf16* wrow = W + (size_t)(nbase + col) * CC;   // B fragment column (= W row)

    for (int kb = 0; kb < CC; kb += 32) {
        // A 16x32 fragment: elems 0..7 @ K=kb+8*half, elems 8..15 @ K=kb+16+8*half
        const __bf16* pa = arow + kb + 8 * half;
        v16bf af = frag_ld(pa, pa + 16);
#pragma unroll
        for (int n = 0; n < 4; ++n) {
            // B 32x16 fragment: elems 0..7 @ K=kb+8*half, elems 8..15 @ K=kb+16+8*half for column nbase+n*16+col
            const __bf16* pb = wrow + (size_t)n * 16 * CC + kb + 8 * half;
            acc[n] = wmma_bf16(af, frag_ld(pb, pb + 16), acc[n]);
        }
    }

    // Epilogue: C layout = (row M = r + 8*half, col N = lane&15) per VGPR r.
#pragma unroll
    for (int n = 0; n < 4; ++n) {
#pragma unroll
        for (int r = 0; r < 8; ++r) {
            const int m    = mbase + r + 8 * half;
            const int ncol = nbase + n * 16 + col;
            const float v  = acc[n][r];
            if (mode == 3) {
                outf[(size_t)m * CC + ncol] = v + bias[ncol];
            } else {
                const int b = m >> 11, t = m & (TT - 1);
                const int h = ncol >> 6, d = ncol & (HD - 1);
                if (mode == 2)
                    outb[(((size_t)b * HH + h) * HD + d) * TT + t] = f2bf(v);
                else
                    outb[(((size_t)b * HH + h) * TT + t) * HD + d] = f2bf(v);
            }
        }
    }
}

// ---------------- flash attention: one wave per (b,h, 16-query tile) -------
// Q,K bf16 [B,H,T,hd]; Vt bf16 [B,H,hd,T]; ctx bf16 [B,T,H,hd]
__global__ __launch_bounds__(256) void attn_kernel(const __bf16* __restrict__ Qb,
                                                   const __bf16* __restrict__ Kb,
                                                   const __bf16* __restrict__ Vt,
                                                   __bf16* __restrict__ ctx) {
    __shared__ __bf16 plds[8][16 * 32];     // per-wave 16x32 P bounce tile

    const int lane  = threadIdx.x & 31;
    const int w     = threadIdx.x >> 5;
    const int gw    = blockIdx.x * 8 + w;       // 0 .. 4095
    const int qt    = gw & (TT / 16 - 1);       // 128 query tiles
    const int bh    = gw >> 7;                  // 0 .. 31
    const int qbase = qt * 16;
    const int col   = lane & 15;
    const int half  = lane >> 4;

    const __bf16* Qp = Qb + (size_t)bh * TT * HD;
    const __bf16* Kp = Kb + (size_t)bh * TT * HD;
    const __bf16* Vp = Vt + (size_t)bh * HD * TT;

    // Q A-fragments (hd chunks 0..31 and 32..63)
    v16bf qa0, qa1;
    {
        const __bf16* r = Qp + (size_t)(qbase + col) * HD + 8 * half;
        qa0 = frag_ld(r, r + 16);
        qa1 = frag_ld(r + 32, r + 48);
    }

    v8f zero = {0.f,0.f,0.f,0.f,0.f,0.f,0.f,0.f};
    v8f o0 = zero, o1 = zero, o2 = zero, o3 = zero;
    float mrun[8], lrun[8];
#pragma unroll
    for (int r = 0; r < 8; ++r) { mrun[r] = -3.0e38f; lrun[r] = 0.f; }

    const int nkt = (qbase + 47) >> 5;          // 32-key tiles with kb <= qbase+15
    for (int j = 0; j < nkt; ++j) {
        const int kb = j * 32;

        // ---- scores S = Q(16x64) . K^T(64x32), two 16x16 column tiles ----
        v8f s0 = zero, s1 = zero;
        {
            const __bf16* k0 = Kp + (size_t)(kb + col) * HD + 8 * half;       // keys kb..kb+15
            const __bf16* k1 = Kp + (size_t)(kb + 16 + col) * HD + 8 * half;  // keys kb+16..kb+31
            s0 = wmma_bf16(qa0, frag_ld(k0,      k0 + 16), s0);
            s0 = wmma_bf16(qa1, frag_ld(k0 + 32, k0 + 48), s0);
            s1 = wmma_bf16(qa0, frag_ld(k1,      k1 + 16), s1);
            s1 = wmma_bf16(qa1, frag_ld(k1 + 32, k1 + 48), s1);
        }

        // ---- scale, causal mask, online softmax ----
        const int key0 = kb + col, key1 = kb + 16 + col;
#pragma unroll
        for (int r = 0; r < 8; ++r) {
            const int q = qbase + r + 8 * half;
            float a = s0[r] * 0.125f; if (key0 > q) a = -3.0e38f;
            float b = s1[r] * 0.125f; if (key1 > q) b = -3.0e38f;
            float t = fmaxf(a, b);
#pragma unroll
            for (int msk = 8; msk; msk >>= 1) t = fmaxf(t, __shfl_xor(t, msk, 32));
            const float mn = fmaxf(mrun[r], t);
            a = __expf(a - mn);
            b = __expf(b - mn);
            float rs = a + b;
#pragma unroll
            for (int msk = 8; msk; msk >>= 1) rs += __shfl_xor(rs, msk, 32);
            const float corr = __expf(mrun[r] - mn);
            lrun[r] = lrun[r] * corr + rs;
            mrun[r] = mn;
            o0[r] *= corr; o1[r] *= corr; o2[r] *= corr; o3[r] *= corr;
            // write P (C layout -> LDS, row-major 16x32 bf16)
            const int row = r + 8 * half;
            plds[w][row * 32 + col]      = f2bf(a);
            plds[w][row * 32 + 16 + col] = f2bf(b);
        }
        asm volatile("s_wait_dscnt 0" ::: "memory");   // same-wave LDS RAW fence

        // ---- reload P as A fragment (16x32) ----
        const __bf16* pp = &plds[w][col * 32 + 8 * half];
        v16bf pa = frag_ld(pp, pp + 16);

        // ---- O += P(16x32) . V(32x64): 4 B-fragments from V^T ----
        const __bf16* v0 = Vp + (size_t)col * TT + kb + 8 * half;
        o0 = wmma_bf16(pa, frag_ld(v0, v0 + 16), o0);
        const __bf16* v1 = v0 + (size_t)16 * TT;
        o1 = wmma_bf16(pa, frag_ld(v1, v1 + 16), o1);
        const __bf16* v2 = v0 + (size_t)32 * TT;
        o2 = wmma_bf16(pa, frag_ld(v2, v2 + 16), o2);
        const __bf16* v3 = v0 + (size_t)48 * TT;
        o3 = wmma_bf16(pa, frag_ld(v3, v3 + 16), o3);
    }

    // ---- normalize, store ctx [B,T,H,hd] bf16 ----
    const int b = bh >> 4, h = bh & 15;
#pragma unroll
    for (int r = 0; r < 8; ++r) {
        const int q = qbase + r + 8 * half;
        const float inv = 1.0f / lrun[r];
        __bf16* dst = ctx + (((size_t)b * TT + q) * HH + h) * HD + col;
        dst[0]  = f2bf(o0[r] * inv);
        dst[16] = f2bf(o1[r] * inv);
        dst[32] = f2bf(o2[r] * inv);
        dst[48] = f2bf(o3[r] * inv);
    }
}

// ---------------- launcher ----------------
extern "C" void kernel_launch(void* const* d_in, const int* in_sizes, int n_in,
                              void* d_out, int out_size, void* d_ws, size_t ws_size,
                              hipStream_t stream) {
    (void)in_sizes; (void)n_in; (void)out_size; (void)ws_size;

    const float* x  = (const float*)d_in[0];
    const float* Wq = (const float*)d_in[1];
    const float* Wk = (const float*)d_in[2];
    const float* Wv = (const float*)d_in[3];
    const float* Wo = (const float*)d_in[4];
    const float* bo = (const float*)d_in[5];
    float* out = (float*)d_out;

    const size_t MB = 1ull << 20;
    char* ws = (char*)d_ws;
    __bf16* xb  = (__bf16*)(ws);            //  8 MiB : x bf16 [M,C]
    __bf16* Wqb = (__bf16*)(ws +  8 * MB);  //  2 MiB
    __bf16* Wkb = (__bf16*)(ws + 10 * MB);  //  2 MiB
    __bf16* Wvb = (__bf16*)(ws + 12 * MB);  //  2 MiB
    __bf16* Wob = (__bf16*)(ws + 14 * MB);  //  2 MiB
    __bf16* Qb  = (__bf16*)(ws + 16 * MB);  //  8 MiB : [B,H,T,hd]
    __bf16* Kb  = (__bf16*)(ws + 24 * MB);  //  8 MiB : [B,H,T,hd]
    __bf16* Vt  = (__bf16*)(ws + 32 * MB);  //  8 MiB : [B,H,hd,T]
    __bf16* ctx = (__bf16*)(ws + 40 * MB);  //  8 MiB : [B,T,H*hd]

    // fp32 -> bf16 converts
    cvt_f32_bf16<<<(MM * CC / 4 + 255) / 256, 256, 0, stream>>>(x,  xb,  MM * CC / 4);
    cvt_f32_bf16<<<(CC * CC / 4 + 255) / 256, 256, 0, stream>>>(Wq, Wqb, CC * CC / 4);
    cvt_f32_bf16<<<(CC * CC / 4 + 255) / 256, 256, 0, stream>>>(Wk, Wkb, CC * CC / 4);
    cvt_f32_bf16<<<(CC * CC / 4 + 255) / 256, 256, 0, stream>>>(Wv, Wvb, CC * CC / 4);
    cvt_f32_bf16<<<(CC * CC / 4 + 255) / 256, 256, 0, stream>>>(Wo, Wob, CC * CC / 4);

    // QKV projections (M=4096, N=1024, K=1024)
    dim3 ggrid(CC / 256, MM / 32);
    gemm_bf16<<<ggrid, 256, 0, stream>>>(xb, Wqb, Qb, nullptr, nullptr, 0);
    gemm_bf16<<<ggrid, 256, 0, stream>>>(xb, Wkb, Kb, nullptr, nullptr, 1);
    gemm_bf16<<<ggrid, 256, 0, stream>>>(xb, Wvb, Vt, nullptr, nullptr, 2);

    // attention: 4096 waves, 8 waves/block
    attn_kernel<<<BB * HH * (TT / 16) / 8, 256, 0, stream>>>(Qb, Kb, Vt, ctx);

    // output projection + bias, fp32 out
    gemm_bf16<<<ggrid, 256, 0, stream>>>(ctx, Wob, nullptr, out, bo, 3);
}
